// DeepSeekV31MoE_47416438948204
// MI455X (gfx1250) — compile-verified, matrix operand order checked by reading, not run on hardware
//
#include <hip/hip_runtime.h>
#include <hip/hip_bf16.h>
#include <math.h>

#define TOK    1024   // B*S
#define HDIM   1024
#define IDIM   512
#define EDIM   64
#define TOPK   8
#define TILE_M 32

typedef __bf16 bf16_t;
typedef __attribute__((ext_vector_type(16))) __bf16    v16bf;
typedef __attribute__((ext_vector_type(8)))  float     v8f;
typedef __attribute__((ext_vector_type(8)))  unsigned  v8u;

#define WMMA_BF16(a, b, c) \
  __builtin_amdgcn_wmma_f32_16x16x32_bf16(false, (a), false, (b), (short)0, (c), false, false)

// Pack two fp32 into a bf16 pair (lo in [15:0], hi in [31:16]) with one v_perm_b32.
__device__ __forceinline__ unsigned pack2bf(float lo, float hi) {
  union { float f; unsigned u; } a, b;
  a.f = lo; b.f = hi;
  return __builtin_amdgcn_perm(b.u + 0x8000u, a.u + 0x8000u, 0x07060302u);
}

// Scalar fp32 -> bf16 (round to nearest even), bit-manipulation only.
__device__ __forceinline__ bf16_t f2bf(float f) {
  union { float f; unsigned u; } c; c.f = f;
  unsigned r = c.u + 0x7FFFu + ((c.u >> 16) & 1u);
  union { unsigned short s; bf16_t h; } o; o.s = (unsigned short)(r >> 16);
  return o.h;
}

// A fragment (16x32 bf16): rows in lanes, K-pairs in VGPRs, from LDS packed uints.
// ISA layout: lane m=lane&15, g=lane>>4; VGPR v holds K pair (v&3)*2 + (v>>2)*16 + g*8.
__device__ __forceinline__ v16bf load_a_frag(const unsigned* base, int ustride, int lane) {
  const int m = lane & 15, g = lane >> 4;
  const unsigned* row = base + m * ustride;
  v8u t;
#pragma unroll
  for (int v = 0; v < 8; ++v)
    t[v] = row[(v & 3) + ((v >> 2) << 3) + (g << 2)];
  return __builtin_bit_cast(v16bf, t);
}

// B fragment (32x16 bf16): N in lanes, K-pairs in VGPRs, loaded straight from
// global fp32 weights (col = W + kbase*RS + n) and converted on the fly.
template <int RS>
__device__ __forceinline__ v16bf load_b_frag_g(const float* __restrict__ col, int g) {
  v8u t;
#pragma unroll
  for (int v = 0; v < 8; ++v) {
    const int k = ((v & 3) << 1) + ((v >> 2) << 4) + (g << 3);
    t[v] = pack2bf(col[(size_t)k * RS], col[(size_t)(k + 1) * RS]);
  }
  return __builtin_bit_cast(v16bf, t);
}

__global__ void init_counts(int* counts) {
  if (threadIdx.x < EDIM) counts[threadIdx.x] = 0;
}

// One block per token, 64 threads (one per expert): fp32 logits, sigmoid, top-8,
// normalized weights scattered into per-expert CSR lists.
__global__ __launch_bounds__(64)
void router_kernel(const float* __restrict__ h, const float* __restrict__ rw,
                   int* __restrict__ counts, int* __restrict__ tokIdx,
                   float* __restrict__ tokW) {
  const int t = blockIdx.x, e = threadIdx.x;
  float acc = 0.f;
  for (int j = 0; j < HDIM; ++j) acc += h[(size_t)t * HDIM + j] * rw[j * EDIM + e];
  const float aff = 1.f / (1.f + __expf(-acc));
  __shared__ float saff[EDIM];
  saff[e] = aff;
  __syncthreads();
  if (e == 0) {
    float sc[TOPK]; int id[TOPK];
    float sum = 0.f;
    for (int k = 0; k < TOPK; ++k) {
      float best = -1.f; int bi = 0;
      for (int x = 0; x < EDIM; ++x) if (saff[x] > best) { best = saff[x]; bi = x; }
      sc[k] = best; id[k] = bi; saff[bi] = -2.f; sum += best;
    }
    const float inv = 1.f / (sum + 1e-9f);
    for (int k = 0; k < TOPK; ++k) {
      const int ie = id[k];
      const int pos = atomicAdd(&counts[ie], 1);
      tokIdx[ie * TOK + pos] = t;
      tokW[ie * TOK + pos]   = sc[k] * inv;
    }
  }
}

// Fused SwiGLU expert: 32-token tile, 512 threads (16 wave32).
// Phase 1: act[32x512] = silu(h*Wg) * (h*Wu) * w  (bf16 tile in LDS, fp32 WMMA acc)
// Phase 2: out[32x1024] (+)= act * Wd
template <bool SHARED>
__global__ __launch_bounds__(512)
void moe_kernel(const float* __restrict__ h,
                const float* __restrict__ Wgate, const float* __restrict__ Wup,
                const float* __restrict__ Wdown,
                const int* __restrict__ counts, const int* __restrict__ tokIdx,
                const float* __restrict__ tokW, float* __restrict__ out) {
  const int tid  = threadIdx.x;
  const int lane = tid & 31;
  const int wv   = tid >> 5;            // wave id 0..15
  const int e    = SHARED ? 0 : blockIdx.y;
  const int nTok = SHARED ? TOK : counts[e];
  const int m0   = blockIdx.x * TILE_M;
  if (m0 >= nTok) return;

  const float* Wg = SHARED ? Wgate : Wgate + (size_t)e * HDIM * IDIM;
  const float* Wu = SHARED ? Wup   : Wup   + (size_t)e * HDIM * IDIM;
  const float* Wd = SHARED ? Wdown : Wdown + (size_t)e * IDIM * HDIM;

  __shared__ unsigned sA[TILE_M * 17];     // 32x32 bf16 chunk as uint pairs, odd stride
  __shared__ unsigned sActU[TILE_M * 257]; // 32x512 bf16 act tile as uint pairs, odd stride
  __shared__ float    sW[TILE_M];
  __shared__ int      sTok[TILE_M];

  if (tid < TILE_M) {
    int gm = m0 + tid, trow = gm; float wt = 1.f;
    if (!SHARED) {
      if (gm < nTok) { trow = tokIdx[e * TOK + gm]; wt = tokW[e * TOK + gm]; }
      else           { trow = 0; wt = 0.f; }
    }
    sTok[tid] = trow; sW[tid] = wt;
  }
  __syncthreads();

  const int g  = lane >> 4;
  const int nc = lane & 15;

  // ---------------- Phase 1: gate & up GEMMs over H in K=32 chunks ----------------
  v8f accG[2][2], accU[2][2];
#pragma unroll
  for (int mt = 0; mt < 2; ++mt)
#pragma unroll
    for (int ns = 0; ns < 2; ++ns)
#pragma unroll
      for (int v = 0; v < 8; ++v) { accG[mt][ns][v] = 0.f; accU[mt][ns][v] = 0.f; }

  const int nb1 = wv * 32;  // this wave's 32-column strip of I

  for (int kc = 0; kc < HDIM / 32; ++kc) {
    __syncthreads();  // previous chunk's A-fragment reads done
    {
      const int m = tid >> 4, kp = tid & 15;
      float lo = 0.f, hi = 0.f;
      if (m0 + m < nTok) {
        const float* hp = h + (size_t)sTok[m] * HDIM + kc * 32 + kp * 2;
        lo = hp[0]; hi = hp[1];
      }
      sA[m * 17 + kp] = pack2bf(lo, hi);
    }
    __syncthreads();
    const v16bf a0 = load_a_frag(sA, 17, lane);
    const v16bf a1 = load_a_frag(sA + 16 * 17, 17, lane);
#pragma unroll
    for (int ns = 0; ns < 2; ++ns) {
      const float* colg = Wg + (size_t)(kc * 32) * IDIM + nb1 + ns * 16 + nc;
      const v16bf bg = load_b_frag_g<IDIM>(colg, g);
      __builtin_prefetch(colg + 32 * IDIM, 0, 1);
      accG[0][ns] = WMMA_BF16(a0, bg, accG[0][ns]);
      accG[1][ns] = WMMA_BF16(a1, bg, accG[1][ns]);
      const float* colu = Wu + (size_t)(kc * 32) * IDIM + nb1 + ns * 16 + nc;
      const v16bf bu = load_b_frag_g<IDIM>(colu, g);
      __builtin_prefetch(colu + 32 * IDIM, 0, 1);
      accU[0][ns] = WMMA_BF16(a0, bu, accU[0][ns]);
      accU[1][ns] = WMMA_BF16(a1, bu, accU[1][ns]);
    }
  }
  __syncthreads();

  // Epilogue: silu(gate) * up * router_weight -> bf16 activation tile in LDS.
  {
    bf16_t* sAct16 = (bf16_t*)sActU;
#pragma unroll
    for (int mt = 0; mt < 2; ++mt)
#pragma unroll
      for (int ns = 0; ns < 2; ++ns)
#pragma unroll
        for (int v = 0; v < 8; ++v) {
          const int m = mt * 16 + v + g * 8;
          const float gt = accG[mt][ns][v], up = accU[mt][ns][v];
          const float a = gt / (1.f + __expf(-gt)) * up * sW[m];
          sAct16[m * 514 + nb1 + ns * 16 + nc] = f2bf(a);
        }
  }
  __syncthreads();

  // ---------------- Phase 2: down GEMM over I in K=32 chunks (no barriers) ----------------
  v8f accD[2][4];
#pragma unroll
  for (int mt = 0; mt < 2; ++mt)
#pragma unroll
    for (int ns = 0; ns < 4; ++ns)
#pragma unroll
      for (int v = 0; v < 8; ++v) accD[mt][ns][v] = 0.f;

  const int nb2 = wv * 64;  // this wave's 64-column strip of H

  for (int kc = 0; kc < IDIM / 32; ++kc) {
    const v16bf a0 = load_a_frag(sActU + kc * 16, 257, lane);
    const v16bf a1 = load_a_frag(sActU + 16 * 257 + kc * 16, 257, lane);
#pragma unroll
    for (int ns = 0; ns < 4; ++ns) {
      const float* cold = Wd + (size_t)(kc * 32) * HDIM + nb2 + ns * 16 + nc;
      const v16bf bd = load_b_frag_g<HDIM>(cold, g);
      __builtin_prefetch(cold + 32 * HDIM, 0, 1);
      accD[0][ns] = WMMA_BF16(a0, bd, accD[0][ns]);
      accD[1][ns] = WMMA_BF16(a1, bd, accD[1][ns]);
    }
  }

  // Epilogue: store (shared expert) or atomic-accumulate (routed experts).
#pragma unroll
  for (int mt = 0; mt < 2; ++mt)
#pragma unroll
    for (int ns = 0; ns < 4; ++ns)
#pragma unroll
      for (int v = 0; v < 8; ++v) {
        const int m = mt * 16 + v + g * 8;
        if (m0 + m < nTok) {
          const size_t off = (size_t)sTok[m] * HDIM + nb2 + ns * 16 + nc;
          if (SHARED) out[off] = accD[mt][ns][v];
          else        atomicAdd(out + off, accD[mt][ns][v]);
        }
      }
}

extern "C" void kernel_launch(void* const* d_in, const int* in_sizes, int n_in,
                              void* d_out, int out_size, void* d_ws, size_t ws_size,
                              hipStream_t stream) {
  (void)in_sizes; (void)n_in; (void)out_size; (void)ws_size;
  const float* h  = (const float*)d_in[0];
  const float* rw = (const float*)d_in[1];
  const float* sg = (const float*)d_in[2];
  const float* su = (const float*)d_in[3];
  const float* sd = (const float*)d_in[4];
  const float* eg = (const float*)d_in[5];
  const float* eu = (const float*)d_in[6];
  const float* ed = (const float*)d_in[7];
  float* out = (float*)d_out;

  char*  ws     = (char*)d_ws;
  int*   counts = (int*)ws;                                               // 64 ints
  int*   tokIdx = (int*)(ws + 256);                                       // E*TOK ints
  float* tokW   = (float*)(ws + 256 + (size_t)EDIM * TOK * sizeof(int));  // E*TOK floats

  init_counts<<<1, 64, 0, stream>>>(counts);
  router_kernel<<<TOK, 64, 0, stream>>>(h, rw, counts, tokIdx, tokW);
  // Shared expert writes every output element first.
  moe_kernel<true><<<dim3(TOK / TILE_M, 1), 512, 0, stream>>>(
      h, sg, su, sd, nullptr, nullptr, nullptr, out);
  // Routed experts accumulate on top (stream-ordered after the shared kernel).
  moe_kernel<false><<<dim3(TOK / TILE_M, EDIM), 512, 0, stream>>>(
      h, eg, eu, ed, counts, tokIdx, tokW, out);
}
